// selfattLayer_27900107555024
// MI455X (gfx1250) — matrix operand order repaired and checked by reading, not compiled, not run on hardware
//
#include <hip/hip_runtime.h>
#include <hip/hip_bf16.h>

typedef __attribute__((ext_vector_type(16))) __bf16 v16bf;
typedef __attribute__((ext_vector_type(8)))  float  v8f;
typedef int v4i __attribute__((vector_size(16)));   // matches builtin param type

struct B32x8 { uint4 lo, hi; };   // 32 bytes = 16 bf16 elements

static __device__ __forceinline__ v16bf as_v16bf(const B32x8& x) {
    return __builtin_bit_cast(v16bf, x);
}
static __device__ __forceinline__ unsigned short bfbits(float v) {
    return __builtin_bit_cast(unsigned short, (__bf16)v);
}

// ---- gfx1250 async global->LDS staging (ASYNCcnt path), guarded ----------
#if defined(__gfx1250__) && __has_builtin(__builtin_amdgcn_global_load_async_to_lds_b128)
#define USE_ASYNC 1
#else
#define USE_ASYNC 0
#endif

static __device__ __forceinline__ void stage_cp16(const __bf16* g, unsigned short* l) {
#if USE_ASYNC
    __builtin_amdgcn_global_load_async_to_lds_b128(
        (__attribute__((address_space(1))) v4i*)(void*)g,
        (__attribute__((address_space(3))) v4i*)(void*)l, 0, 0);
#else
    *(uint4*)l = *(const uint4*)g;
#endif
}
static __device__ __forceinline__ void stage_wait() {
#if USE_ASYNC
#if __has_builtin(__builtin_amdgcn_s_wait_asynccnt)
    __builtin_amdgcn_s_wait_asynccnt(0);
#else
    asm volatile("s_wait_asynccnt 0x0" ::: "memory");
#endif
#endif
}

// ---------------------------------------------------------------------------
// K0a: x f32 -> bf16 (one pass)
// ---------------------------------------------------------------------------
__global__ void cvt_x_kernel(const float* __restrict__ x, __bf16* __restrict__ xb, int n) {
    int i = (blockIdx.x * blockDim.x + threadIdx.x) * 4;
    if (i >= n) return;
    float4 v = *(const float4*)(x + i);
    unsigned int lo = (unsigned int)bfbits(v.x) | ((unsigned int)bfbits(v.y) << 16);
    unsigned int hi = (unsigned int)bfbits(v.z) | ((unsigned int)bfbits(v.w) << 16);
    uint2 o; o.x = lo; o.y = hi;
    *(uint2*)(xb + i) = o;
}

// ---------------------------------------------------------------------------
// K0b: W [K][N] f32 -> WT [N][K] bf16 (tiny)
// ---------------------------------------------------------------------------
__global__ void transpose_w_kernel(const float* __restrict__ W,
                                   __bf16* __restrict__ WT, int K, int N) {
    int i = blockIdx.x * blockDim.x + threadIdx.x;
    if (i >= K * N) return;
    int n = i / K;
    int k = i - n * K;
    WT[n * K + k] = (__bf16)W[k * N + n];
}

// ---------------------------------------------------------------------------
// K1: projections. xb [16384][256] bf16 -> f,g bf16 [16384][64] row-major,
//     h stored transposed: hT bf16 [4][256][4096] (LDS bounce for coalescing).
// ---------------------------------------------------------------------------
__global__ __launch_bounds__(128) void proj_kernel(
    const __bf16* __restrict__ xb,
    const __bf16* __restrict__ WfT,   // [64][256]
    const __bf16* __restrict__ WgT,   // [64][256]
    const __bf16* __restrict__ WhT,   // [256][256]
    const float*  __restrict__ bf_,
    const float*  __restrict__ bg_,
    const float*  __restrict__ bh_,
    __bf16* __restrict__ fbuf,        // [16384][64]
    __bf16* __restrict__ gbuf,        // [16384][64]
    __bf16* __restrict__ hT)          // [4][256][4096]
{
    __shared__ __align__(16) unsigned short tlds[4][256];   // 512 B per wave

    const int lane  = threadIdx.x & 31;
    const int wv    = threadIdx.x >> 5;
    const int job   = blockIdx.x * 4 + wv;   // 0 .. 24575
    const int mtile = job / 24;
    const int j     = job - mtile * 24;

    const __bf16* WT;
    const float*  bias;
    int ntile;
    if (j < 4)       { WT = WfT; bias = bf_; ntile = j;     }
    else if (j < 8)  { WT = WgT; bias = bg_; ntile = j - 4; }
    else             { WT = WhT; bias = bh_; ntile = j - 8; }

    const int l15 = lane & 15;
    const int kh  = lane >> 4;

    const __bf16* xrow = xb + (size_t)(mtile * 16 + l15) * 256;
    const __bf16* wrow = WT + (size_t)(ntile * 16 + l15) * 256;

    v8f acc = {};
    #pragma unroll
    for (int kc = 0; kc < 8; ++kc) {
        B32x8 ta;   // A: chunks K = kc*32+kh*8 and kc*32+16+kh*8
        ta.lo = *(const uint4*)(xrow + kc * 32 + kh * 8);
        ta.hi = *(const uint4*)(xrow + kc * 32 + 16 + kh * 8);
        B32x8 tb;   // B: lane n=l15, chunks K = kc*32+kh*8 and kc*32+16+kh*8
        tb.lo = *(const uint4*)(wrow + kc * 32 + kh * 8);
        tb.hi = *(const uint4*)(wrow + kc * 32 + 16 + kh * 8);
        acc = __builtin_amdgcn_wmma_f32_16x16x32_bf16(
                  false, as_v16bf(ta), false, as_v16bf(tb), (short)0, acc, false, false);
    }

    const int   ncol = ntile * 16 + l15;
    const float bv   = bias[ncol];

    if (j < 8) {
        __bf16* dst = (j < 4) ? fbuf : gbuf;
        #pragma unroll
        for (int r = 0; r < 8; ++r) {
            int grow = mtile * 16 + r + 8 * kh;
            dst[(size_t)grow * 64 + ncol] = (__bf16)(acc[r] + bv);
        }
    } else {
        // bounce 16x16 tile through LDS so hT stores are 16B contiguous
        unsigned short* tw = &tlds[wv][0];
        #pragma unroll
        for (int r = 0; r < 8; ++r)
            tw[(r + 8 * kh) * 16 + l15] = bfbits(acc[r] + bv);
        __builtin_amdgcn_wave_barrier();
        int batch = (mtile * 16) >> 12;          // 16-row tile never straddles a batch
        int nin0  = (mtile * 16) & 4095;
        unsigned int w[4];
        #pragma unroll
        for (int i = 0; i < 4; ++i) {
            unsigned int lo = tw[(kh * 8 + 2 * i) * 16 + l15];
            unsigned int hi = tw[(kh * 8 + 2 * i + 1) * 16 + l15];
            w[i] = lo | (hi << 16);
        }
        uint4 q; q.x = w[0]; q.y = w[1]; q.z = w[2]; q.w = w[3];
        __bf16* drow = hT + (size_t)batch * (256 * 4096) + (size_t)ncol * 4096 + nin0 + kh * 8;
        *(uint4*)drow = q;
    }
}

// ---------------------------------------------------------------------------
// K2: fused flash attention + residual. 8 waves/block share double-buffered
// 64-key slabs staged asynchronously into LDS; each wave owns 16 queries
// x all 256 channels.
// ---------------------------------------------------------------------------
__global__ __launch_bounds__(256) void attn_kernel(
    const __bf16* __restrict__ fbuf,  // [4][4096][64]
    const __bf16* __restrict__ gbuf,  // [4][4096][64]
    const __bf16* __restrict__ hT,    // [4][256][4096]
    const float*  __restrict__ x,     // [4][4096][256]
    const float*  __restrict__ scalep,
    float* __restrict__ out)          // [4][4096][256]
{
    __shared__ __align__(16) unsigned short f_lds[2][64 * 64];    // 16 KB [buf][key][d]
    __shared__ __align__(16) unsigned short h_lds[2][256 * 64];   // 64 KB [buf][c][key]
    __shared__ __align__(16) unsigned short p_lds[8][16 * 64];    // 16 KB per-wave P

    const int tid   = threadIdx.x;
    const int lane  = tid & 31;
    const int wv    = tid >> 5;
    const int b     = blockIdx.x >> 5;                 // 32 blocks per batch
    const int qtile = (blockIdx.x & 31) * 8 + wv;      // 0..255 within batch
    const int l15   = lane & 15;
    const int kh    = lane >> 4;

    const __bf16* fb = fbuf + (size_t)b * 4096 * 64;
    const __bf16* gb = gbuf + (size_t)b * 4096 * 64;
    const __bf16* hb = hT   + (size_t)b * 256 * 4096;

    // g A-fragments: two K=32 chunks of d=64, reused for every key slab
    const __bf16* grow_p = gb + (size_t)(qtile * 16 + l15) * 64;
    v16bf ga0, ga1;
    {
        B32x8 t;
        t.lo = *(const uint4*)(grow_p + kh * 8);
        t.hi = *(const uint4*)(grow_p + 16 + kh * 8);
        ga0 = as_v16bf(t);
        t.lo = *(const uint4*)(grow_p + 32 + kh * 8);
        t.hi = *(const uint4*)(grow_p + 48 + kh * 8);
        ga1 = as_v16bf(t);
    }

    v8f oacc[16];
    #pragma unroll
    for (int t = 0; t < 16; ++t) oacc[t] = (v8f){};
    float mrun[8], lrun[8];
    #pragma unroll
    for (int r = 0; r < 8; ++r) { mrun[r] = -__builtin_inff(); lrun[r] = 0.f; }

    unsigned short* pw = &p_lds[wv][0];

    // per-thread staging addresses: f slab 32B/thread, h slab 128B/thread
    const int frow = tid >> 2, fq4 = tid & 3;

    // prologue: stage slab 0 into buffer 0
    {
        const __bf16* fs = fb + (size_t)frow * 64 + fq4 * 16;
        stage_cp16(fs,     &f_lds[0][frow * 64 + fq4 * 16]);
        stage_cp16(fs + 8, &f_lds[0][frow * 64 + fq4 * 16 + 8]);
        const __bf16* hs = hb + (size_t)tid * 4096;
        #pragma unroll
        for (int i = 0; i < 8; ++i) stage_cp16(hs + i * 8, &h_lds[0][tid * 64 + i * 8]);
    }
    stage_wait();
    __syncthreads();

    #pragma unroll 2
    for (int kb = 0; kb < 64; ++kb) {
        const int p = kb & 1;

        // ---- async-stage next slab into the other buffer ----
        if (kb + 1 < 64) {
            const int nk0 = (kb + 1) * 64;
            const __bf16* fs = fb + (size_t)(nk0 + frow) * 64 + fq4 * 16;
            stage_cp16(fs,     &f_lds[p ^ 1][frow * 64 + fq4 * 16]);
            stage_cp16(fs + 8, &f_lds[p ^ 1][frow * 64 + fq4 * 16 + 8]);
            const __bf16* hs = hb + (size_t)tid * 4096 + nk0;
            #pragma unroll
            for (int i = 0; i < 8; ++i) stage_cp16(hs + i * 8, &h_lds[p ^ 1][tid * 64 + i * 8]);
        }

        // ---- S = g @ f^T : four 16x16 f32 tiles over 64 keys ----
        v8f sfr[4];
        #pragma unroll
        for (int jt = 0; jt < 4; ++jt) sfr[jt] = (v8f){};
        #pragma unroll
        for (int kc = 0; kc < 2; ++kc) {
            v16bf ga = kc ? ga1 : ga0;
            #pragma unroll
            for (int jt = 0; jt < 4; ++jt) {
                const unsigned short* fp = &f_lds[p][(jt * 16 + l15) * 64 + kc * 32 + kh * 8];
                B32x8 t;
                t.lo = *(const uint4*)(fp);
                t.hi = *(const uint4*)(fp + 16);
                sfr[jt] = __builtin_amdgcn_wmma_f32_16x16x32_bf16(
                              false, ga, false, as_v16bf(t), (short)0, sfr[jt], false, false);
            }
        }

        // ---- online softmax (row m = r + 8*kh lives in this lane's half) ----
        #pragma unroll
        for (int r = 0; r < 8; ++r) {
            float bm = fmaxf(fmaxf(sfr[0][r], sfr[1][r]), fmaxf(sfr[2][r], sfr[3][r]));
            #pragma unroll
            for (int m = 1; m < 16; m <<= 1) bm = fmaxf(bm, __shfl_xor(bm, m, 32));
            float mn    = fmaxf(mrun[r], bm);
            float alpha = __expf(mrun[r] - mn);
            mrun[r] = mn;
            float e0 = __expf(sfr[0][r] - mn);
            float e1 = __expf(sfr[1][r] - mn);
            float e2 = __expf(sfr[2][r] - mn);
            float e3 = __expf(sfr[3][r] - mn);
            float rs = (e0 + e1) + (e2 + e3);
            #pragma unroll
            for (int m = 1; m < 16; m <<= 1) rs += __shfl_xor(rs, m, 32);
            lrun[r] = lrun[r] * alpha + rs;
            #pragma unroll
            for (int t = 0; t < 16; ++t) oacc[t][r] *= alpha;
            int m0 = (r + 8 * kh) * 64 + l15;
            pw[m0]      = bfbits(e0);
            pw[m0 + 16] = bfbits(e1);
            pw[m0 + 32] = bfbits(e2);
            pw[m0 + 48] = bfbits(e3);
        }
        __builtin_amdgcn_wave_barrier();   // same-wave LDS is in-order

        // ---- P A-fragments (16x64 -> two K=32 chunks) from wave-private LDS ----
        v16bf pa0, pa1;
        {
            B32x8 t;
            t.lo = *(const uint4*)&pw[l15 * 64 + kh * 8];
            t.hi = *(const uint4*)&pw[l15 * 64 + 16 + kh * 8];
            pa0 = as_v16bf(t);
            t.lo = *(const uint4*)&pw[l15 * 64 + 32 + kh * 8];
            t.hi = *(const uint4*)&pw[l15 * 64 + 48 + kh * 8];
            pa1 = as_v16bf(t);
        }

        // ---- o += P @ h_slab over 16 channel tiles x 2 K-chunks ----
        #pragma unroll
        for (int t = 0; t < 16; ++t) {
            const unsigned short* hp = &h_lds[p][(t * 16 + l15) * 64];
            B32x8 b0;
            b0.lo = *(const uint4*)(hp + kh * 8);
            b0.hi = *(const uint4*)(hp + 16 + kh * 8);
            oacc[t] = __builtin_amdgcn_wmma_f32_16x16x32_bf16(
                          false, pa0, false, as_v16bf(b0), (short)0, oacc[t], false, false);
            B32x8 b1;
            b1.lo = *(const uint4*)(hp + 32 + kh * 8);
            b1.hi = *(const uint4*)(hp + 48 + kh * 8);
            oacc[t] = __builtin_amdgcn_wmma_f32_16x16x32_bf16(
                          false, pa1, false, as_v16bf(b1), (short)0, oacc[t], false, false);
        }

        stage_wait();      // our async loads into the other buffer are done
        __syncthreads();   // everyone done reading buf p / writing buf p^1
    }

    // ---- epilogue: y = scale * (o / l) + x ----
    const float scale = *scalep;
    float inv[8];
    #pragma unroll
    for (int r = 0; r < 8; ++r) inv[r] = 1.0f / lrun[r];
    #pragma unroll
    for (int t = 0; t < 16; ++t) {
        #pragma unroll
        for (int r = 0; r < 8; ++r) {
            int    q   = qtile * 16 + r + 8 * kh;
            size_t idx = ((size_t)b * 4096 + q) * 256 + t * 16 + l15;
            out[idx] = scale * (oacc[t][r] * inv[r]) + x[idx];
        }
    }
}

// ---------------------------------------------------------------------------
extern "C" void kernel_launch(void* const* d_in, const int* in_sizes, int n_in,
                              void* d_out, int out_size, void* d_ws, size_t ws_size,
                              hipStream_t stream) {
    const float* x     = (const float*)d_in[0];
    const float* Wf    = (const float*)d_in[1];
    const float* bf_   = (const float*)d_in[2];
    const float* Wg    = (const float*)d_in[3];
    const float* bg_   = (const float*)d_in[4];
    const float* Wh    = (const float*)d_in[5];
    const float* bh_   = (const float*)d_in[6];
    const float* scale = (const float*)d_in[7];
    float*       out   = (float*)d_out;

    char* ws = (char*)d_ws;
    __bf16* WfT = (__bf16*)(ws + 0);                       // 32 KB
    __bf16* WgT = (__bf16*)(ws + 32768);                   // 32 KB
    __bf16* WhT = (__bf16*)(ws + 65536);                   // 128 KB
    __bf16* fb  = (__bf16*)(ws + 196608);                  // 2 MB
    __bf16* gb  = (__bf16*)(ws + 196608 + 2097152);        // 2 MB
    __bf16* hT  = (__bf16*)(ws + 196608 + 2 * 2097152);    // 8 MB
    __bf16* xbf = (__bf16*)(ws + 196608 + 2 * 2097152 + 8388608); // 8 MB

    cvt_x_kernel<<<4096, 256, 0, stream>>>(x, xbf, 4 * 4096 * 256);
    transpose_w_kernel<<<(256 * 64  + 255) / 256, 256, 0, stream>>>(Wf, WfT, 256, 64);
    transpose_w_kernel<<<(256 * 64  + 255) / 256, 256, 0, stream>>>(Wg, WgT, 256, 64);
    transpose_w_kernel<<<(256 * 256 + 255) / 256, 256, 0, stream>>>(Wh, WhT, 256, 256);

    // 1024 M-tiles * 24 N-tiles = 24576 wave-jobs, 4 waves/block
    proj_kernel<<<6144, 128, 0, stream>>>(xbf, WfT, WgT, WhT, bf_, bg_, bh_, fb, gb, hT);

    // 4 batches * 32 blocks, 8 waves/block, 16 queries/wave
    attn_kernel<<<128, 256, 0, stream>>>(fb, gb, hT, x, scale, out);
}
